// HierarchicalGNN_34428457844763
// MI455X (gfx1250) — matrix operand order repaired and checked by reading, not compiled, not run on hardware
//
#include <hip/hip_runtime.h>
#include <hip/hip_bf16.h>
#include <math.h>

#define NNODES 50000
#define NEDGES 800000
#define E2 (NEDGES + NNODES)
#define NGRAPH 64
#define FNODE 64
#define FGRAPH 16
#define HID 256
#define HEADS 4
#define DH 64
#define NLAYERS 3
#define NCLS 10
#define NEG_SLOPE 0.2f
#define BN_EPS 1e-5f

typedef _Float16 v16h __attribute__((ext_vector_type(16)));
typedef float v8f __attribute__((ext_vector_type(8)));

union Frag16 { unsigned int u[8]; v16h h; };
union PackH2 { _Float16 h[2]; unsigned int u; };

__device__ __forceinline__ unsigned int f2ord(float x) {
  unsigned int b = __float_as_uint(x);
  return (b & 0x80000000u) ? ~b : (b | 0x80000000u);
}
__device__ __forceinline__ float ord2f(unsigned int u) {
  unsigned int b = (u & 0x80000000u) ? (u & 0x7FFFFFFFu) : ~u;
  return __uint_as_float(b);
}

// ---------------------------------------------------------------------------
// WMMA GEMM: C[M x Nc] = A[M x K] @ B[K x Nc] (+ bias) (opt tanh)
// K % 32 == 0, Nc % 64 == 0. Block: 256 thr = 8 waves; tile 128x64.
// A rows are CLAMPED (not zeroed) when out of range: garbage only reaches
// accumulator rows that are never stored, so no divergent load path.
// All 4 B fragments are materialized before the WMMA sequence so the DS
// waits retire progressively instead of a full dscnt-0 drain per WMMA.
// ---------------------------------------------------------------------------
__global__ __launch_bounds__(256) void k_gemm_wmma(
    const float* __restrict__ A, const float* __restrict__ B,
    const float* __restrict__ bias, float* __restrict__ C,
    int M, int K, int Nc, int act)
{
  __shared__ _Float16 sA[128][34];   // [row][k], row stride 68B (4B aligned)
  __shared__ _Float16 sB[64][34];    // [col n][k]
  const int tid  = threadIdx.x;
  const int lane = tid & 31;
  const int wave = tid >> 5;
  const int row0 = blockIdx.y * 128;
  const int col0 = blockIdx.x * 64;
  const int g    = lane >> 4;        // lane half
  const int lm   = lane & 15;

  const v8f vz = {0.f,0.f,0.f,0.f,0.f,0.f,0.f,0.f};
  v8f acc[4] = {vz, vz, vz, vz};

  for (int k0 = 0; k0 < K; k0 += 32) {
    // ---- A tile 128x32: float4 loads, clamped row (branchless) ----
    #pragma unroll
    for (int i = 0; i < 4; ++i) {
      int grp = tid + i * 256;          // 0..1023 float4 groups
      int r   = grp >> 3;               // 0..127
      int k4  = (grp & 7) << 2;         // 0,4,...,28
      int gr  = row0 + r;
      int gra = (gr < M) ? gr : (M - 1);
      float4 v = *(const float4*)&A[(size_t)gra * K + k0 + k4];
      PackH2 p0, p1;
      p0.h[0] = (_Float16)v.x; p0.h[1] = (_Float16)v.y;
      p1.h[0] = (_Float16)v.z; p1.h[1] = (_Float16)v.w;
      *(unsigned int*)&sA[r][k4]     = p0.u;
      *(unsigned int*)&sA[r][k4 + 2] = p1.u;
    }
    // ---- B tile 32x64: float4 across n, transpose into sB[n][k] ----
    #pragma unroll
    for (int i = 0; i < 2; ++i) {
      int grp = tid + i * 256;          // 0..511 float4 groups
      int kk  = grp >> 4;               // 0..31
      int n4  = (grp & 15) << 2;        // 0,4,...,60
      float4 v = *(const float4*)&B[(size_t)(k0 + kk) * Nc + col0 + n4];
      sB[n4 + 0][kk] = (_Float16)v.x;
      sB[n4 + 1][kk] = (_Float16)v.y;
      sB[n4 + 2][kk] = (_Float16)v.z;
      sB[n4 + 3][kk] = (_Float16)v.w;
    }
    __syncthreads();

    // A fragment: lane m = lm; dword v -> kpair (v<4? v : v+4) + 4g
    Frag16 fa;
    #pragma unroll
    for (int v = 0; v < 8; ++v) {
      int kp = (v < 4 ? v : v + 4) + 4 * g;
      fa.u[v] = *(const unsigned int*)&sA[wave * 16 + lm][2 * kp];
    }
    // All 4 B fragments (distinct registers -> progressive dscnt waits)
    Frag16 fb[4];
    #pragma unroll
    for (int t = 0; t < 4; ++t)
      #pragma unroll
      for (int v = 0; v < 8; ++v)
        fb[t].u[v] = *(const unsigned int*)&sB[t * 16 + lm][2 * ((v < 4 ? v : v + 4) + 4 * g)];

    #pragma unroll
    for (int t = 0; t < 4; ++t)
      acc[t] = __builtin_amdgcn_wmma_f32_16x16x32_f16(
          false, fa.h, false, fb[t].h, (short)0, acc[t], false, false);
    __syncthreads();
  }

  // C layout: VGPR j -> row j (lanes 0-15) / j+8 (lanes 16-31), col = lm
  #pragma unroll
  for (int t = 0; t < 4; ++t) {
    #pragma unroll
    for (int j = 0; j < 8; ++j) {
      int gr = row0 + wave * 16 + j + 8 * g;
      int gc = col0 + t * 16 + lm;
      if (gr < M) {
        float v = acc[t][j];
        if (bias) v += bias[gc];
        if (act == 1) v = tanhf(v);
        C[(size_t)gr * Nc + gc] = v;
      }
    }
  }
}

// ---------------------------------------------------------------------------
// GAT attention pieces
// ---------------------------------------------------------------------------
__global__ void k_alpha(const float* __restrict__ hh,
                        const float* __restrict__ asrc,
                        const float* __restrict__ adst,
                        float* __restrict__ oas, float* __restrict__ oad)
{
  int id = blockIdx.x * blockDim.x + threadIdx.x;
  if (id >= NNODES * HEADS) return;
  int n = id >> 2, h = id & 3;
  const float* base = hh + (size_t)n * HID + h * DH;
  const float* ws = asrc + h * DH;
  const float* wd = adst + h * DH;
  float s = 0.f, d = 0.f;
  #pragma unroll 4
  for (int i = 0; i < DH; i += 4) {
    float4 v = *(const float4*)&base[i];
    float4 a = *(const float4*)&ws[i];
    float4 b = *(const float4*)&wd[i];
    s += v.x * a.x + v.y * a.y + v.z * a.z + v.w * a.w;
    d += v.x * b.x + v.y * b.y + v.z * b.z + v.w * b.w;
  }
  oas[id] = s; oad[id] = d;
}

__global__ void k_edge_max(const int* __restrict__ ei,
                           const float* __restrict__ oas,
                           const float* __restrict__ oad,
                           float* __restrict__ ebuf,
                           unsigned int* __restrict__ mbits)
{
  int i = blockIdx.x * blockDim.x + threadIdx.x;
  if (i >= E2) return;
  int s, d;
  if (i < NEDGES) { s = ei[i]; d = ei[NEDGES + i]; }
  else            { s = i - NEDGES; d = s; }
  float4 as = *(const float4*)&oas[s * 4];
  float4 ad = *(const float4*)&oad[d * 4];
  float e0 = as.x + ad.x, e1 = as.y + ad.y, e2 = as.z + ad.z, e3 = as.w + ad.w;
  e0 = e0 > 0.f ? e0 : NEG_SLOPE * e0;
  e1 = e1 > 0.f ? e1 : NEG_SLOPE * e1;
  e2 = e2 > 0.f ? e2 : NEG_SLOPE * e2;
  e3 = e3 > 0.f ? e3 : NEG_SLOPE * e3;
  float4 ev = {e0, e1, e2, e3};
  *(float4*)&ebuf[(size_t)i * 4] = ev;
  atomicMax(&mbits[d * 4 + 0], f2ord(e0));
  atomicMax(&mbits[d * 4 + 1], f2ord(e1));
  atomicMax(&mbits[d * 4 + 2], f2ord(e2));
  atomicMax(&mbits[d * 4 + 3], f2ord(e3));
}

__global__ void k_edge_exp(const int* __restrict__ ei,
                           const unsigned int* __restrict__ mbits,
                           float* __restrict__ ebuf,
                           float* __restrict__ denom)
{
  int i = blockIdx.x * blockDim.x + threadIdx.x;
  if (i >= E2) return;
  int d = (i < NEDGES) ? ei[NEDGES + i] : (i - NEDGES);
  float4 ev = *(const float4*)&ebuf[(size_t)i * 4];
  uint4 mb = *(const uint4*)&mbits[d * 4];
  float p0 = __expf(ev.x - ord2f(mb.x));
  float p1 = __expf(ev.y - ord2f(mb.y));
  float p2 = __expf(ev.z - ord2f(mb.z));
  float p3 = __expf(ev.w - ord2f(mb.w));
  float4 pv = {p0, p1, p2, p3};
  *(float4*)&ebuf[(size_t)i * 4] = pv;
  atomicAdd(&denom[d * 4 + 0], p0);
  atomicAdd(&denom[d * 4 + 1], p1);
  atomicAdd(&denom[d * 4 + 2], p2);
  atomicAdd(&denom[d * 4 + 3], p3);
}

__global__ void k_init_bias(float* __restrict__ out, const float* __restrict__ b)
{
  int id = blockIdx.x * blockDim.x + threadIdx.x;
  if (id >= NNODES * HID) return;
  out[id] = b[id & (HID - 1)];
}

// one thread per (edge, 8-float chunk); c = part*8..part*8+7, head = part>>3
__global__ void k_aggregate(const int* __restrict__ ei,
                            const float* __restrict__ hh,
                            const float* __restrict__ ebuf,
                            const float* __restrict__ denom,
                            float* __restrict__ out)
{
  long long gid = (long long)blockIdx.x * blockDim.x + threadIdx.x;
  if (gid >= (long long)E2 * 32) return;
  int i = (int)(gid >> 5);
  int part = (int)(gid & 31);
  int s, d;
  if (i < NEDGES) { s = ei[i]; d = ei[NEDGES + i]; }
  else            { s = i - NEDGES; d = s; }
  int h = part >> 3;
  float w = ebuf[(size_t)i * 4 + h] / denom[d * 4 + h];
  const float4* src = (const float4*)(hh + (size_t)s * HID + part * 8);
  float4 v0 = src[0], v1 = src[1];
  float* dst = out + (size_t)d * HID + part * 8;
  atomicAdd(&dst[0], v0.x * w);
  atomicAdd(&dst[1], v0.y * w);
  atomicAdd(&dst[2], v0.z * w);
  atomicAdd(&dst[3], v0.w * w);
  atomicAdd(&dst[4], v1.x * w);
  atomicAdd(&dst[5], v1.y * w);
  atomicAdd(&dst[6], v1.z * w);
  atomicAdd(&dst[7], v1.w * w);
}

// ---------------------------------------------------------------------------
// BatchNorm (training-mode stats) + ELU
// ---------------------------------------------------------------------------
__global__ void k_bn_stats(const float* __restrict__ h, float* __restrict__ sums)
{
  int c = threadIdx.x;  // blockDim.x == HID
  float s = 0.f, s2 = 0.f;
  for (int n = blockIdx.x; n < NNODES; n += gridDim.x) {
    float v = h[(size_t)n * HID + c];
    s += v; s2 += v * v;
  }
  atomicAdd(&sums[c], s);
  atomicAdd(&sums[HID + c], s2);
}

__global__ void k_bn_apply(float* __restrict__ h, const float* __restrict__ sums,
                           const float* __restrict__ g, const float* __restrict__ b)
{
  int id = blockIdx.x * blockDim.x + threadIdx.x;
  if (id >= NNODES * HID) return;
  int c = id & (HID - 1);
  float mu = sums[c] * (1.f / NNODES);
  float var = sums[HID + c] * (1.f / NNODES) - mu * mu;
  float x = (h[id] - mu) * rsqrtf(var + BN_EPS) * g[c] + b[c];
  h[id] = x > 0.f ? x : (__expf(x) - 1.f);
}

// ---------------------------------------------------------------------------
// Global attention pooling
// ---------------------------------------------------------------------------
__global__ void k_attn(const float* __restrict__ s1, const float* __restrict__ w2,
                       const float* __restrict__ b2, float* __restrict__ attn,
                       unsigned int* __restrict__ maxbits)
{
  int n = blockIdx.x * blockDim.x + threadIdx.x;
  if (n >= NNODES) return;
  const float* r = s1 + (size_t)n * 128;
  float a = b2[0];
  #pragma unroll 8
  for (int i = 0; i < 128; i += 4) {
    float4 v = *(const float4*)&r[i];
    float4 wv = *(const float4*)&w2[i];
    a += v.x * wv.x + v.y * wv.y + v.z * wv.z + v.w * wv.w;
  }
  attn[n] = a;
  atomicMax(maxbits, f2ord(a));
}

__global__ void k_attn_exp(const float* __restrict__ attn,
                           const unsigned int* __restrict__ maxbits,
                           float* __restrict__ p, float* __restrict__ total)
{
  int n = blockIdx.x * blockDim.x + threadIdx.x;
  if (n >= NNODES) return;
  float v = __expf(attn[n] - ord2f(*maxbits));
  p[n] = v;
  atomicAdd(total, v);
}

__global__ void k_pool(const float* __restrict__ h, const float* __restrict__ p,
                       const float* __restrict__ total, const int* __restrict__ batch,
                       float* __restrict__ pooled)
{
  long long gid = (long long)blockIdx.x * blockDim.x + threadIdx.x;
  if (gid >= (long long)NNODES * 32) return;
  int n = (int)(gid >> 5);
  int part = (int)(gid & 31);
  float w = p[n] / *total;
  int g = batch[n];
  const float4* src = (const float4*)(h + (size_t)n * HID + part * 8);
  float4 v0 = src[0], v1 = src[1];
  float* dst = pooled + (size_t)g * HID + part * 8;
  atomicAdd(&dst[0], v0.x * w);
  atomicAdd(&dst[1], v0.y * w);
  atomicAdd(&dst[2], v0.z * w);
  atomicAdd(&dst[3], v0.w * w);
  atomicAdd(&dst[4], v1.x * w);
  atomicAdd(&dst[5], v1.y * w);
  atomicAdd(&dst[6], v1.z * w);
  atomicAdd(&dst[7], v1.w * w);
}

// ---------------------------------------------------------------------------
// Classifier
// ---------------------------------------------------------------------------
__global__ void k_cls1(const float* __restrict__ pooled, const float* __restrict__ gf,
                       const float* __restrict__ w, const float* __restrict__ b,
                       float* __restrict__ z)
{
  int id = blockIdx.x * blockDim.x + threadIdx.x;
  if (id >= NGRAPH * HID) return;
  int g = id >> 8, j = id & (HID - 1);
  float a = b[j];
  const float* pg = pooled + (size_t)g * HID;
  for (int k = 0; k < HID; ++k) a += pg[k] * w[(size_t)k * HID + j];
  const float* gfe = gf + g * FGRAPH;
  for (int k = 0; k < FGRAPH; ++k) a += gfe[k] * w[(size_t)(HID + k) * HID + j];
  z[id] = a > 0.f ? a : 0.f;
}

__global__ void k_cls2(const float* __restrict__ z, const float* __restrict__ w,
                       const float* __restrict__ b, float* __restrict__ out)
{
  int id = blockIdx.x * blockDim.x + threadIdx.x;
  if (id >= NGRAPH * NCLS) return;
  int g = id / NCLS, c = id % NCLS;
  float a = b[c];
  const float* zg = z + (size_t)g * HID;
  for (int k = 0; k < HID; ++k) a += zg[k] * w[(size_t)k * NCLS + c];
  out[id] = a;
}

// ---------------------------------------------------------------------------
extern "C" void kernel_launch(void* const* d_in, const int* in_sizes, int n_in,
                              void* d_out, int out_size, void* d_ws, size_t ws_size,
                              hipStream_t stream)
{
  (void)in_sizes; (void)n_in; (void)out_size; (void)ws_size;
  const float* x     = (const float*)d_in[0];
  const int*   ei    = (const int*)d_in[1];
  const int*   batch = (const int*)d_in[2];
  const float* gf    = (const float*)d_in[3];
  const float* encW  = (const float*)d_in[4];
  const float* encB  = (const float*)d_in[5];
  const float* gatW  = (const float*)d_in[6];
  const float* gatAs = (const float*)d_in[7];
  const float* gatAd = (const float*)d_in[8];
  const float* gatB  = (const float*)d_in[9];
  const float* bnG   = (const float*)d_in[10];
  const float* bnB   = (const float*)d_in[11];
  const float* pw1   = (const float*)d_in[12];
  const float* pb1   = (const float*)d_in[13];
  const float* pw2   = (const float*)d_in[14];
  const float* pb2   = (const float*)d_in[15];
  const float* cw1   = (const float*)d_in[16];
  const float* cb1   = (const float*)d_in[17];
  const float* cw2   = (const float*)d_in[18];
  const float* cb2   = (const float*)d_in[19];
  float* out = (float*)d_out;

  char* ws = (char*)d_ws;
  size_t off = 0;
  auto alloc = [&](size_t bytes) -> char* {
    char* p = ws + off;
    off += (bytes + 255) & ~(size_t)255;
    return p;
  };
  float*        bufA   = (float*)alloc((size_t)NNODES * HID * 4);   // h (51.2 MB)
  float*        bufB   = (float*)alloc((size_t)NNODES * HID * 4);   // hh / s1
  float*        ebuf   = (float*)alloc((size_t)E2 * HEADS * 4);     // 13.6 MB
  unsigned int* mbits  = (unsigned int*)alloc((size_t)NNODES * HEADS * 4);
  float*        denom  = (float*)alloc((size_t)NNODES * HEADS * 4);
  float*        oas    = (float*)alloc((size_t)NNODES * HEADS * 4);
  float*        oad    = (float*)alloc((size_t)NNODES * HEADS * 4);
  float*        bnsum  = (float*)alloc(2 * HID * 4);
  float*        attn   = (float*)alloc((size_t)NNODES * 4);
  float*        pbuf   = (float*)alloc((size_t)NNODES * 4);
  float*        scal   = (float*)alloc(256);                        // [0]=maxbits [1]=total
  float*        pooled = (float*)alloc((size_t)NGRAPH * HID * 4);
  float*        zbuf   = (float*)alloc((size_t)NGRAPH * HID * 4);

  dim3 blk(256);
  const int nhBlocks   = (NNODES * HEADS + 255) / 256;
  const int eBlocks    = (E2 + 255) / 256;
  const int aggBlocks  = (int)(((long long)E2 * 32 + 255) / 256);
  const int nhidBlocks = (NNODES * HID + 255) / 256;
  const int nBlocks    = (NNODES + 255) / 256;

  // node encoder: h = x @ enc_W + enc_b
  {
    dim3 grid(HID / 64, (NNODES + 127) / 128);
    k_gemm_wmma<<<grid, blk, 0, stream>>>(x, encW, encB, bufA, NNODES, FNODE, HID, 0);
  }

  for (int l = 0; l < NLAYERS; ++l) {
    dim3 grid(HID / 64, (NNODES + 127) / 128);
    k_gemm_wmma<<<grid, blk, 0, stream>>>(bufA, gatW + (size_t)l * HID * HID,
                                          nullptr, bufB, NNODES, HID, HID, 0);
    k_alpha<<<nhBlocks, blk, 0, stream>>>(bufB, gatAs + l * HEADS * DH,
                                          gatAd + l * HEADS * DH, oas, oad);
    hipMemsetAsync(mbits, 0, (size_t)NNODES * HEADS * 4, stream);
    k_edge_max<<<eBlocks, blk, 0, stream>>>(ei, oas, oad, ebuf, mbits);
    hipMemsetAsync(denom, 0, (size_t)NNODES * HEADS * 4, stream);
    k_edge_exp<<<eBlocks, blk, 0, stream>>>(ei, mbits, ebuf, denom);
    k_init_bias<<<nhidBlocks, blk, 0, stream>>>(bufA, gatB + l * HID);
    k_aggregate<<<aggBlocks, blk, 0, stream>>>(ei, bufB, ebuf, denom, bufA);
    hipMemsetAsync(bnsum, 0, 2 * HID * 4, stream);
    k_bn_stats<<<512, HID, 0, stream>>>(bufA, bnsum);
    k_bn_apply<<<nhidBlocks, blk, 0, stream>>>(bufA, bnsum, bnG + l * HID, bnB + l * HID);
  }

  // attention pooling: s1 = tanh(h @ pool_w1 + b1)
  {
    dim3 grid(128 / 64, (NNODES + 127) / 128);
    k_gemm_wmma<<<grid, blk, 0, stream>>>(bufA, pw1, pb1, bufB, NNODES, HID, 128, 1);
  }
  hipMemsetAsync(scal, 0, 256, stream);
  k_attn<<<nBlocks, blk, 0, stream>>>(bufB, pw2, pb2, attn, (unsigned int*)scal);
  k_attn_exp<<<nBlocks, blk, 0, stream>>>(attn, (const unsigned int*)scal, pbuf, scal + 1);
  hipMemsetAsync(pooled, 0, (size_t)NGRAPH * HID * 4, stream);
  {
    int poolBlocks = (int)(((long long)NNODES * 32 + 255) / 256);
    k_pool<<<poolBlocks, blk, 0, stream>>>(bufA, pbuf, scal + 1, batch, pooled);
  }
  k_cls1<<<(NGRAPH * HID + 255) / 256, blk, 0, stream>>>(pooled, gf, cw1, cb1, zbuf);
  k_cls2<<<(NGRAPH * NCLS + 255) / 256, blk, 0, stream>>>(zbuf, cw2, cb2, out);
}
